// NER_CRF_Classifier_28080496181436
// MI455X (gfx1250) — compile-verified, matrix operand order checked by reading, not run on hardware
//
#include <hip/hip_runtime.h>
#include <math.h>

// ---------------------------------------------------------------------------
// BERT-base + linear head + CRF log-likelihood, MI455X (gfx1250, wave32).
// GEMMs/attention use v_wmma_f32_16x16x32_f16 with f32 accumulation.
// LDS tiles are stored FRAGMENT-MAJOR (lane's 16 halves contiguous, 32B
// aligned) so every WMMA operand load is exactly two ds_load_b128.
// GEMM is register-blocked 2x in M: 16 WMMAs per 20 ds_load_b128.
// Workspace (floats): x,q,k,v,ctx,tmp (8192*768 each), ffh (2048*3072),
// logits (8192*13), E (16*512*13)  => ~177 MB.
// ---------------------------------------------------------------------------

typedef __attribute__((ext_vector_type(16))) _Float16 v16h;
typedef __attribute__((ext_vector_type(2)))  _Float16 h2;
typedef __attribute__((ext_vector_type(8)))  float    v8f;

#define DEV __device__ __forceinline__

// Forward map (ISA 7.12.2, 16-bit A 16x32 / B 32x16): element (mn, k) lives at
//   lane = (mn & 15) | ((k & 8) << 1)
//   h    = (k & 7)   | ((k & 16) >> 1)
DEV int frag_k(int h, int lane) {
  int v = h >> 1, p = h & 1;
  return ((v & 4) ? 16 : 0) + ((lane & 16) ? 8 : 0) + ((v & 3) << 1) + p;
}
DEV int inv_lane(int mn, int k) { return (mn & 15) | ((k & 8) << 1); }
DEV int inv_h(int k) { return (k & 7) | ((k & 16) >> 1); }

DEV v8f wmma_f16(v16h a, v16h b, v8f c) {
  return __builtin_amdgcn_wmma_f32_16x16x32_f16(false, a, false, b,
                                                (short)0, c, false, false);
}

// Block-wide sum over 256 threads (8 waves of 32).
DEV float block_sum256(float v, float* red) {
  #pragma unroll
  for (int d = 16; d >= 1; d >>= 1) v += __shfl_xor(v, d, 32);
  int lane = threadIdx.x & 31, w = threadIdx.x >> 5;
  if (lane == 0) red[w] = v;
  __syncthreads();
  float r = (threadIdx.x < 8) ? red[threadIdx.x] : 0.f;
  if (w == 0) {
    #pragma unroll
    for (int d = 4; d >= 1; d >>= 1) r += __shfl_xor(r, d, 32);
    if (lane == 0) red[0] = r;
  }
  __syncthreads();
  float out = red[0];
  __syncthreads();
  return out;
}

// ---------------------------------------------------------------------------
// GEMM: C[M,N] = A[M,K] @ W[K,N] + bias, optional exact-GELU epilogue.
// BM=256, BN=128, BK=32; 256 threads = 8 waves; wave w -> rows 32w..32w+31
// (2 A frags), 16 C fragments across 128 cols. M%256==0, N%128==0, K%32==0.
// ---------------------------------------------------------------------------
__global__ __launch_bounds__(256) void k_gemm(
    const float* __restrict__ A, const float* __restrict__ W,
    const float* __restrict__ bias, float* __restrict__ C,
    int M, int N, int K, int gelu) {
  (void)M;
  __shared__ __align__(32) _Float16 AsF[16 * 32 * 16];  // [wavetile][lane][h]
  __shared__ __align__(32) _Float16 WsF[8 * 32 * 16];   // [ntile][lane][h]
  int tid = threadIdx.x;
  int lane = tid & 31, wave = tid >> 5;
  int bn = blockIdx.x * 128, bm = blockIdx.y * 256;
  int n0 = lane & 15;
  int hi8 = (lane & 16) ? 8 : 0;

  v8f acc[2][8];
  #pragma unroll
  for (int p = 0; p < 2; p++)
    #pragma unroll
    for (int i = 0; i < 8; i++)
      #pragma unroll
      for (int r = 0; r < 8; r++) acc[p][i][r] = 0.f;

  for (int k0 = 0; k0 < K; k0 += 32) {
    __syncthreads();
    // Stage A tile 256x32 as k-pairs: float2 global load -> b32 LDS store.
    #pragma unroll
    for (int it = 0; it < 16; it++) {
      int idx = it * 256 + tid;          // 0..4095
      int rowr = idx >> 4, kp = idx & 15;
      int k = kp * 2;
      float2 av = *(const float2*)&A[(size_t)(bm + rowr) * K + k0 + k];
      h2 p; p.x = (_Float16)av.x; p.y = (_Float16)av.y;
      int off = (((rowr >> 4) * 32 + inv_lane(rowr, k)) << 4) + inv_h(k);
      *(h2*)&AsF[off] = p;
    }
    // Stage W tile 32x128 as k-pairs (n fast for coalescing).
    #pragma unroll
    for (int it = 0; it < 8; it++) {
      int idx = it * 256 + tid;          // 0..2047
      int kp = idx >> 7, n = idx & 127;
      int k = kp * 2;
      h2 p;
      p.x = (_Float16)W[(size_t)(k0 + k) * N + bn + n];
      p.y = (_Float16)W[(size_t)(k0 + k + 1) * N + bn + n];
      int off = (((n >> 4) * 32 + inv_lane(n, k)) << 4) + inv_h(k);
      *(h2*)&WsF[off] = p;
    }
    __syncthreads();

    v16h af0 = *(const v16h*)&AsF[((wave * 2 + 0) * 32 + lane) << 4];
    v16h af1 = *(const v16h*)&AsF[((wave * 2 + 1) * 32 + lane) << 4];
    #pragma unroll
    for (int nt = 0; nt < 8; nt++) {
      v16h bf = *(const v16h*)&WsF[(nt * 32 + lane) << 4];
      acc[0][nt] = wmma_f16(af0, bf, acc[0][nt]);
      acc[1][nt] = wmma_f16(af1, bf, acc[1][nt]);
    }
  }

  #pragma unroll
  for (int p = 0; p < 2; p++) {
    #pragma unroll
    for (int nt = 0; nt < 8; nt++) {
      int n = nt * 16 + n0;
      float bv = bias[bn + n];
      #pragma unroll
      for (int r = 0; r < 8; r++) {
        int m = wave * 32 + p * 16 + r + hi8;
        float v = acc[p][nt][r] + bv;
        if (gelu) v = 0.5f * v * (1.f + erff(v * 0.70710678f));
        C[(size_t)(bm + m) * N + bn + n] = v;
      }
    }
  }
}

// ---------------------------------------------------------------------------
// Flash attention: grid.x = B*NH (192), grid.y = L/128 (4); 8 waves/block,
// wave owns 16 query rows. Per 32-key step the block cooperatively stages
// K^T and V tiles into fragment-major LDS shared by all waves.
// ---------------------------------------------------------------------------
__global__ __launch_bounds__(256) void k_attention(
    const float* __restrict__ Q, const float* __restrict__ K,
    const float* __restrict__ V, const int* __restrict__ am,
    float* __restrict__ O) {
  __shared__ __align__(32) _Float16 KsF[4 * 32 * 16];  // f = sub*2 + dh-slice
  __shared__ __align__(32) _Float16 VsF[4 * 32 * 16];  // f = dh col tile
  __shared__ __align__(32) _Float16 PsF[8 * 32 * 16];  // per-wave P (A frag)
  int tid = threadIdx.x;
  int lane = tid & 31, wave = tid >> 5;
  int b = blockIdx.x / 12, h = blockIdx.x % 12;
  int q0 = blockIdx.y * 128 + wave * 16;
  size_t base = ((size_t)b * 512) * 768 + (size_t)h * 64;
  int n = lane & 15;
  int hi8 = (lane & 16) ? 8 : 0;

  // Q fragments: 8 x float2 per fragment (k pairs are h-contiguous).
  v16h aq0, aq1;
  {
    size_t row = base + (size_t)(q0 + n) * 768;
    #pragma unroll
    for (int hp = 0; hp < 8; hp++) {
      int k = frag_k(2 * hp, lane);
      float2 q2 = *(const float2*)&Q[row + k];
      aq0[2 * hp] = (_Float16)q2.x; aq0[2 * hp + 1] = (_Float16)q2.y;
      float2 q3 = *(const float2*)&Q[row + 32 + k];
      aq1[2 * hp] = (_Float16)q3.x; aq1[2 * hp + 1] = (_Float16)q3.y;
    }
  }

  float mst[8], lst[8];
  #pragma unroll
  for (int r = 0; r < 8; r++) { mst[r] = -3.0e38f; lst[r] = 0.f; }
  v8f oc[4];
  #pragma unroll
  for (int c = 0; c < 4; c++)
    #pragma unroll
    for (int r = 0; r < 8; r++) oc[c][r] = 0.f;

  for (int j = 0; j < 16; j++) {
    int key0 = j * 32;
    __syncthreads();  // previous-iteration readers done before overwrite
    // Stage K tile (32 keys x 64 dh) as dh-pairs, dh fast (coalesced).
    #pragma unroll
    for (int it = 0; it < 4; it++) {
      int idx = it * 256 + tid;          // 0..1023
      int kk = idx >> 5, pi = idx & 31;
      int d = pi * 2;
      float2 kv = *(const float2*)&K[base + (size_t)(key0 + kk) * 768 + d];
      h2 p; p.x = (_Float16)kv.x; p.y = (_Float16)kv.y;
      int s = d >> 5, kd = d & 31;
      int f = (kk >> 4) * 2 + s;
      int off = ((f * 32 + inv_lane(kk, kd)) << 4) + inv_h(kd);
      *(h2*)&KsF[off] = p;
    }
    // Stage V tile (32 keys x 64 dh) as key-pairs, dh fast (coalesced).
    #pragma unroll
    for (int it = 0; it < 4; it++) {
      int idx = it * 256 + tid;          // 0..1023
      int kkp = idx >> 6, d = idx & 63;
      int kk = kkp * 2;
      h2 p;
      p.x = (_Float16)V[base + (size_t)(key0 + kk) * 768 + d];
      p.y = (_Float16)V[base + (size_t)(key0 + kk + 1) * 768 + d];
      int f = d >> 4;
      int off = ((f * 32 + inv_lane(d, kk)) << 4) + inv_h(kk);
      *(h2*)&VsF[off] = p;
    }
    __syncthreads();

    v8f sm[2];
    #pragma unroll
    for (int sub = 0; sub < 2; sub++) {
      v16h bk0 = *(const v16h*)&KsF[((sub * 2 + 0) * 32 + lane) << 4];
      v16h bk1 = *(const v16h*)&KsF[((sub * 2 + 1) * 32 + lane) << 4];
      v8f s;
      #pragma unroll
      for (int r = 0; r < 8; r++) s[r] = 0.f;
      s = wmma_f16(aq0, bk0, s);
      s = wmma_f16(aq1, bk1, s);
      sm[sub] = s;
    }
    float e0 = (1.f - (float)am[b * 512 + key0 + n]) * -1e9f;
    float e1 = (1.f - (float)am[b * 512 + key0 + 16 + n]) * -1e9f;
    #pragma unroll
    for (int r = 0; r < 8; r++) {
      float a0 = sm[0][r] * 0.125f + e0;
      float a1 = sm[1][r] * 0.125f + e1;
      float mx = fmaxf(a0, a1);
      #pragma unroll
      for (int d = 8; d >= 1; d >>= 1) mx = fmaxf(mx, __shfl_xor(mx, d, 32));
      float mn = fmaxf(mst[r], mx);
      float al = __expf(mst[r] - mn);
      float p0 = __expf(a0 - mn);
      float p1 = __expf(a1 - mn);
      float rs = p0 + p1;
      #pragma unroll
      for (int d = 8; d >= 1; d >>= 1) rs += __shfl_xor(rs, d, 32);
      lst[r] = lst[r] * al + rs;
      mst[r] = mn;
      #pragma unroll
      for (int c = 0; c < 4; c++) oc[c][r] *= al;
      // Scatter P straight into A-fragment layout (row = m, col = key k).
      int m = r + hi8;
      int al0 = inv_lane(m, n);           // k = n       (p0)
      PsF[((wave * 32 + al0) << 4) + inv_h(n)] = (_Float16)p0;
      PsF[((wave * 32 + al0) << 4) + inv_h(16 + n)] = (_Float16)p1;
    }
    asm volatile("s_wait_dscnt 0x0" ::: "memory");  // cross-lane P exchange
    v16h ap = *(const v16h*)&PsF[(wave * 32 + lane) << 4];
    #pragma unroll
    for (int c = 0; c < 4; c++) {
      v16h bv = *(const v16h*)&VsF[(c * 32 + lane) << 4];
      oc[c] = wmma_f16(ap, bv, oc[c]);
    }
  }

  #pragma unroll
  for (int c = 0; c < 4; c++) {
    #pragma unroll
    for (int r = 0; r < 8; r++) {
      int mm = r + hi8;
      float inv = 1.f / fmaxf(lst[r], 1e-20f);
      O[base + (size_t)(q0 + mm) * 768 + c * 16 + n] = oc[c][r] * inv;
    }
  }
}

// ---------------------------------------------------------------------------
// Embedding sum + LayerNorm. One block (256 thr) per token; 3 cols/thread.
// ---------------------------------------------------------------------------
__global__ __launch_bounds__(256) void k_embed_ln(
    const int* __restrict__ ids, const int* __restrict__ tt,
    const float* __restrict__ we, const float* __restrict__ pe,
    const float* __restrict__ te, const float* __restrict__ g,
    const float* __restrict__ bb, float* __restrict__ X) {
  __shared__ float red[8];
  int tok = blockIdx.x;
  int l = tok & 511;
  int id = ids[tok], ty = tt[tok];
  float v[3]; float s = 0.f;
  #pragma unroll
  for (int i = 0; i < 3; i++) {
    int c = threadIdx.x + i * 256;
    v[i] = we[(size_t)id * 768 + c] + pe[(size_t)l * 768 + c] +
           te[(size_t)ty * 768 + c];
    s += v[i];
  }
  float mean = block_sum256(s, red) * (1.f / 768.f);
  float s2 = 0.f;
  #pragma unroll
  for (int i = 0; i < 3; i++) { float d = v[i] - mean; s2 += d * d; }
  float var = block_sum256(s2, red) * (1.f / 768.f);
  float inv = rsqrtf(var + 1e-12f);
  #pragma unroll
  for (int i = 0; i < 3; i++) {
    int c = threadIdx.x + i * 256;
    X[(size_t)tok * 768 + c] = (v[i] - mean) * inv * g[c] + bb[c];
  }
}

// X = LayerNorm(X + R) * g + b, in place. One block per token.
__global__ __launch_bounds__(256) void k_ln_residual(
    float* __restrict__ X, const float* __restrict__ R,
    const float* __restrict__ g, const float* __restrict__ bb) {
  __shared__ float red[8];
  int tok = blockIdx.x;
  float v[3]; float s = 0.f;
  #pragma unroll
  for (int i = 0; i < 3; i++) {
    int c = threadIdx.x + i * 256;
    v[i] = X[(size_t)tok * 768 + c] + R[(size_t)tok * 768 + c];
    s += v[i];
  }
  float mean = block_sum256(s, red) * (1.f / 768.f);
  float s2 = 0.f;
  #pragma unroll
  for (int i = 0; i < 3; i++) { float d = v[i] - mean; s2 += d * d; }
  float var = block_sum256(s2, red) * (1.f / 768.f);
  float inv = rsqrtf(var + 1e-12f);
  #pragma unroll
  for (int i = 0; i < 3; i++) {
    int c = threadIdx.x + i * 256;
    X[(size_t)tok * 768 + c] = (v[i] - mean) * inv * g[c] + bb[c];
  }
}

// logits[tok,t] = x[tok,:] @ lin_W[:,t] + lin_b[t]   (T=13, tiny)
__global__ void k_logits(const float* __restrict__ X, const float* __restrict__ W,
                         const float* __restrict__ b, float* __restrict__ out) {
  int idx = blockIdx.x * blockDim.x + threadIdx.x;
  if (idx >= 8192 * 13) return;
  int tok = idx / 13, t = idx - tok * 13;
  const float* xr = X + (size_t)tok * 768;
  float s = b[t];
  for (int k = 0; k < 768; k++) s += xr[k] * W[k * 13 + t];
  out[idx] = s;
}

// ---------------------------------------------------------------------------
// CRF: gather kept-token logits (stable, zero-padded), gold-path score,
// forward-algorithm logZ, output = sum_b(score - logZ). One block, 256 thr:
// thread = (b in 0..15) x (t in 0..15, t<13 active).
// ---------------------------------------------------------------------------
__global__ __launch_bounds__(256) void k_crf(
    const float* __restrict__ logits, const unsigned char* __restrict__ cmask,
    const unsigned char* __restrict__ smask, const int* __restrict__ tags,
    const float* __restrict__ start, const float* __restrict__ endv,
    const float* __restrict__ trans, float* __restrict__ E,
    float* __restrict__ out) {
  __shared__ float shT[13][13], shS[13], shE[13];
  __shared__ float aA[16][13], aB[16][13];
  __shared__ float sc[16], lz[16];
  int tid = threadIdx.x;
  if (tid < 169) shT[tid / 13][tid % 13] = trans[tid];
  if (tid < 13) { shS[tid] = start[tid]; shE[tid] = endv[tid]; }
  __syncthreads();
  int b = tid >> 4, t = tid & 15;

  // Gather: kept tokens first (original order), zero-pad tail.
  if (t < 13) {
    int cnt = 0;
    for (int i = 0; i < 512; i++) {
      if (cmask[b * 512 + i]) {
        E[((size_t)b * 512 + cnt) * 13 + t] = logits[((size_t)b * 512 + i) * 13 + t];
        cnt++;
      }
    }
    for (int i = cnt; i < 512; i++) E[((size_t)b * 512 + i) * 13 + t] = 0.f;
  }
  __syncthreads();

  // Numerator: gold path score.
  if (t == 0) {
    const int* tg = tags + b * 512;
    float score = shS[tg[0]] + E[((size_t)b * 512) * 13 + tg[0]];
    int nkept = 0;
    for (int i = 0; i < 512; i++) nkept += smask[b * 512 + i] ? 1 : 0;
    for (int i = 1; i < 512; i++) {
      float mf = smask[b * 512 + i] ? 1.f : 0.f;
      score += (shT[tg[i - 1]][tg[i]] + E[((size_t)b * 512 + i) * 13 + tg[i]]) * mf;
    }
    score += shE[tg[nkept - 1]];
    sc[b] = score;
  }

  // Denominator: forward algorithm.
  if (t < 13) aA[b][t] = shS[t] + E[((size_t)b * 512) * 13 + t];
  __syncthreads();
  for (int i = 1; i < 512; i++) {
    if (t < 13) {
      if (smask[b * 512 + i]) {
        float mx = -3e38f;
        for (int tp = 0; tp < 13; tp++) mx = fmaxf(mx, aA[b][tp] + shT[tp][t]);
        float s = 0.f;
        for (int tp = 0; tp < 13; tp++) s += expf(aA[b][tp] + shT[tp][t] - mx);
        aB[b][t] = mx + logf(s) + E[((size_t)b * 512 + i) * 13 + t];
      } else {
        aB[b][t] = aA[b][t];
      }
    }
    __syncthreads();
    if (t < 13) aA[b][t] = aB[b][t];
    __syncthreads();
  }
  if (t == 0) {
    float mx = -3e38f;
    for (int tp = 0; tp < 13; tp++) mx = fmaxf(mx, aA[b][tp] + shE[tp]);
    float s = 0.f;
    for (int tp = 0; tp < 13; tp++) s += expf(aA[b][tp] + shE[tp] - mx);
    lz[b] = mx + logf(s);
  }
  __syncthreads();
  if (tid == 0) {
    float tot = 0.f;
    for (int bi = 0; bi < 16; bi++) tot += sc[bi] - lz[bi];
    out[0] = tot;
  }
}

// ---------------------------------------------------------------------------
// Host orchestration.
// ---------------------------------------------------------------------------
extern "C" void kernel_launch(void* const* d_in, const int* in_sizes, int n_in,
                              void* d_out, int out_size, void* d_ws, size_t ws_size,
                              hipStream_t stream) {
  (void)in_sizes; (void)n_in; (void)out_size; (void)ws_size;

  const int*           input_ids       = (const int*)d_in[0];
  const int*           attention_mask  = (const int*)d_in[1];
  const int*           token_type_ids  = (const int*)d_in[2];
  const unsigned char* crf_mask        = (const unsigned char*)d_in[3];  // jax bool = i8
  const unsigned char* sorted_crf_mask = (const unsigned char*)d_in[4];
  const int*           tag_sequence    = (const int*)d_in[5];
  const float* word_emb = (const float*)d_in[6];
  const float* pos_emb  = (const float*)d_in[7];
  const float* type_emb = (const float*)d_in[8];
  const float* emb_ln_g = (const float*)d_in[9];
  const float* emb_ln_b = (const float*)d_in[10];
  const float* lWq = (const float*)d_in[11];
  const float* lbq = (const float*)d_in[12];
  const float* lWk = (const float*)d_in[13];
  const float* lbk = (const float*)d_in[14];
  const float* lWv = (const float*)d_in[15];
  const float* lbv = (const float*)d_in[16];
  const float* lWo = (const float*)d_in[17];
  const float* lbo = (const float*)d_in[18];
  const float* lln1_g = (const float*)d_in[19];
  const float* lln1_b = (const float*)d_in[20];
  const float* lW1 = (const float*)d_in[21];
  const float* lb1 = (const float*)d_in[22];
  const float* lW2 = (const float*)d_in[23];
  const float* lb2 = (const float*)d_in[24];
  const float* lln2_g = (const float*)d_in[25];
  const float* lln2_b = (const float*)d_in[26];
  const float* lin_W = (const float*)d_in[27];
  const float* lin_b = (const float*)d_in[28];
  const float* crf_start = (const float*)d_in[29];
  const float* crf_end   = (const float*)d_in[30];
  const float* crf_trans = (const float*)d_in[31];

  const size_t TOK = 8192;           // B*L
  float* ws  = (float*)d_ws;
  float* x   = ws;                   // [8192,768]
  float* qb  = x   + TOK * 768;
  float* kb  = qb  + TOK * 768;
  float* vb  = kb  + TOK * 768;
  float* ctx = vb  + TOK * 768;
  float* tmp = ctx + TOK * 768;
  float* ffh = tmp + TOK * 768;      // [2048,3072] chunk buffer
  float* lg  = ffh + (size_t)2048 * 3072;  // [8192,13]
  float* Eb  = lg  + TOK * 13;       // [16,512,13]

  k_embed_ln<<<8192, 256, 0, stream>>>(input_ids, token_type_ids, word_emb,
                                       pos_emb, type_emb, emb_ln_g, emb_ln_b, x);

  for (int l = 0; l < 12; l++) {
    size_t oHH = (size_t)l * 768 * 768;
    size_t oH  = (size_t)l * 768;
    size_t oHF = (size_t)l * 768 * 3072;
    size_t oF  = (size_t)l * 3072;

    k_gemm<<<dim3(6, 32), 256, 0, stream>>>(x, lWq + oHH, lbq + oH, qb, 8192, 768, 768, 0);
    k_gemm<<<dim3(6, 32), 256, 0, stream>>>(x, lWk + oHH, lbk + oH, kb, 8192, 768, 768, 0);
    k_gemm<<<dim3(6, 32), 256, 0, stream>>>(x, lWv + oHH, lbv + oH, vb, 8192, 768, 768, 0);
    k_attention<<<dim3(192, 4), 256, 0, stream>>>(qb, kb, vb, attention_mask, ctx);
    k_gemm<<<dim3(6, 32), 256, 0, stream>>>(ctx, lWo + oHH, lbo + oH, tmp, 8192, 768, 768, 0);
    k_ln_residual<<<8192, 256, 0, stream>>>(x, tmp, lln1_g + oH, lln1_b + oH);

    for (int c = 0; c < 4; c++) {
      k_gemm<<<dim3(24, 8), 256, 0, stream>>>(x + (size_t)c * 2048 * 768,
                                              lW1 + oHF, lb1 + oF, ffh,
                                              2048, 3072, 768, 1);
      k_gemm<<<dim3(6, 8), 256, 0, stream>>>(ffh, lW2 + oHF, lb2 + oH,
                                             tmp + (size_t)c * 2048 * 768,
                                             2048, 768, 3072, 0);
    }
    k_ln_residual<<<8192, 256, 0, stream>>>(x, tmp, lln2_g + oH, lln2_b + oH);
  }

  k_logits<<<(8192 * 13 + 255) / 256, 256, 0, stream>>>(x, lin_W, lin_b, lg);
  k_crf<<<1, 256, 0, stream>>>(lg, crf_mask, sorted_crf_mask, tag_sequence,
                               crf_start, crf_end, crf_trans, Eb, (float*)d_out);
}
